// SwinTransformerBlock3D_46935402610971
// MI455X (gfx1250) — matrix operand order repaired and checked by reading, not compiled, not run on hardware
//
#include <hip/hip_runtime.h>
#include <hip/hip_bf16.h>

typedef __attribute__((ext_vector_type(16))) __bf16 v16bf;
typedef __attribute__((ext_vector_type(8)))  __bf16 v8bf;
typedef __attribute__((ext_vector_type(8)))  float  v8f;

union V16U { v16bf v; v8bf h[2]; };

#define TOKENS  43904   // B * 28^3 = 2 * 21952
#define TOKS_B  21952
#define NW      343     // tokens per window
#define NPAD    352     // 343 padded to 22*16
#define CCH     128
#define NHEAD   4
#define HD      32
#define HIDDEN  512
#define NWIN    128     // B * (28/7)^3
#define MTILES  2744    // 43904/16
#define SCALE_Q 0.17677669529663687f

// ---------------------------------------------------------------- WMMA core
static __device__ __forceinline__ v8f wmma_bf16(v16bf a, v16bf b, v8f c) {
  // (neg_a, A, neg_b, B, c_mod, C, reuse_a, reuse_b)
  return __builtin_amdgcn_wmma_f32_16x16x32_bf16(false, a, false, b, (short)0, c,
                                                 false, false);
}

// One wave computes a 16x64 strip: 4 consecutive 16x16 N-tiles sharing the
// A fragment (4x WMMA per A-load).  Y = X(M,KC) @ W(N,KC)^T, bf16 in, f32 acc.
template <int KC>
static __device__ __forceinline__ void gemm_strip4(const __bf16* __restrict__ X,
                                                   const __bf16* __restrict__ W,
                                                   size_t mtile, int ntile0,
                                                   v8f acc[4]) {
  const int lane = threadIdx.x & 31;
  const int mr = lane & 15;
  const int kh = lane >> 4;
  const __bf16* arow = X + (mtile * 16 + (size_t)mr) * KC;
  const __bf16* brow = W + ((size_t)ntile0 * 16 + mr) * KC;
#pragma unroll 4
  for (int kk = 0; kk < KC / 32; ++kk) {
    V16U au;
    au.h[0] = *(const v8bf*)(arow + kk * 32 + kh * 16);
    au.h[1] = *(const v8bf*)(arow + kk * 32 + kh * 16 + 8);
#pragma unroll
    for (int t = 0; t < 4; ++t) {
      v16bf bf = *(const v16bf*)(brow + (size_t)t * 16 * KC + kk * 32 + kh * 16);
      acc[t] = wmma_bf16(au.v, bf, acc[t]);
    }
  }
}

static __device__ __forceinline__ float wave_sum(float v) {
#pragma unroll
  for (int off = 16; off > 0; off >>= 1) v += __shfl_xor(v, off, 32);
  return v;
}
static __device__ __forceinline__ float wave_max(float v) {
#pragma unroll
  for (int off = 16; off > 0; off >>= 1) v = fmaxf(v, __shfl_xor(v, off, 32));
  return v;
}

// ---------------------------------------------------------------- converters
__global__ void cvt_f32_bf16_kernel(const float* __restrict__ s,
                                    __bf16* __restrict__ d, int n) {
  int i = blockIdx.x * blockDim.x + threadIdx.x;
  if (i < n) d[i] = (__bf16)s[i];
}

// ------------------------------------------------- LN1 + shift + window part
__global__ void ln1_partition_kernel(const float* __restrict__ x,
                                     const float* __restrict__ w,
                                     const float* __restrict__ b,
                                     __bf16* __restrict__ out) {
  int wave = threadIdx.x >> 5, lane = threadIdx.x & 31;
  size_t tg = (size_t)blockIdx.x * 4 + wave;          // window token id
  size_t win = tg / NW;
  int tok = (int)(tg % NW);
  int bb = (int)(win >> 6), rem = (int)(win & 63);
  int wh = rem >> 4, ww = (rem >> 2) & 3, wd = rem & 3;
  int i = tok / 49, j = (tok / 7) % 7, kz = tok % 7;
  int ah = (wh * 7 + i + 3) % 28;                      // roll(-3) gather
  int aw = (ww * 7 + j + 3) % 28;
  int ad = (wd * 7 + kz + 3) % 28;
  size_t src = ((size_t)ah * 28 + aw) * 28 + ad;
  const float* row = x + ((size_t)bb * TOKS_B + src) * CCH;
  float v[4], s = 0.f, sq = 0.f;
#pragma unroll
  for (int t = 0; t < 4; ++t) {
    v[t] = row[lane + 32 * t];
    s += v[t];
    sq += v[t] * v[t];
  }
  s = wave_sum(s);
  sq = wave_sum(sq);
  float mean = s * (1.0f / 128.0f);
  float var = sq * (1.0f / 128.0f) - mean * mean;
  float rstd = rsqrtf(var + 1e-5f);
#pragma unroll
  for (int t = 0; t < 4; ++t) {
    int c = lane + 32 * t;
    out[tg * CCH + c] = (__bf16)((v[t] - mean) * rstd * w[c] + b[c]);
  }
}

// ------------------------------------------------------------------ LN2
__global__ void ln2_kernel(const float* __restrict__ x2,
                           const float* __restrict__ w,
                           const float* __restrict__ b,
                           __bf16* __restrict__ out) {
  int wave = threadIdx.x >> 5, lane = threadIdx.x & 31;
  size_t tg = (size_t)blockIdx.x * 4 + wave;
  const float* row = x2 + tg * CCH;
  float v[4], s = 0.f, sq = 0.f;
#pragma unroll
  for (int t = 0; t < 4; ++t) {
    v[t] = row[lane + 32 * t];
    s += v[t];
    sq += v[t] * v[t];
  }
  s = wave_sum(s);
  sq = wave_sum(sq);
  float mean = s * (1.0f / 128.0f);
  float var = sq * (1.0f / 128.0f) - mean * mean;
  float rstd = rsqrtf(var + 1e-5f);
#pragma unroll
  for (int t = 0; t < 4; ++t) {
    int c = lane + 32 * t;
    out[tg * CCH + c] = (__bf16)((v[t] - mean) * rstd * w[c] + b[c]);
  }
}

// -------------------------------------------------------------- QKV GEMM
// Y(43904,384) = xw @ qkv_w^T + b; scatter into (win, head, tok, hd) bf16,
// q pre-scaled by HD^-0.5.  One wave: 16 rows x 64 cols.
__global__ void qkv_gemm_kernel(const __bf16* __restrict__ xw,
                                const __bf16* __restrict__ wq,
                                const float* __restrict__ qkv_b,
                                __bf16* __restrict__ qb,
                                __bf16* __restrict__ kb,
                                __bf16* __restrict__ vb) {
  const int NT4 = 384 / 64;                       // 6 strips
  int wave = threadIdx.x >> 5, lane = threadIdx.x & 31;
  size_t tile = (size_t)blockIdx.x * 4 + wave;
  int nt4 = (int)(tile % NT4);
  size_t mt = tile / NT4;
  if (mt >= MTILES) return;
  int n = lane & 15, kh = lane >> 4;
  v8f acc[4];
#pragma unroll
  for (int t = 0; t < 4; ++t) {
    float bv = qkv_b[(nt4 * 4 + t) * 16 + n];
#pragma unroll
    for (int r = 0; r < 8; ++r) acc[t][r] = bv;
  }
  gemm_strip4<CCH>(xw, wq, mt, nt4 * 4, acc);
#pragma unroll
  for (int t = 0; t < 4; ++t) {
    int col = (nt4 * 4 + t) * 16 + n;
    int mat = col >> 7, cc = col & 127, hh = cc >> 5, d = cc & 31;
    __bf16* dst = (mat == 0) ? qb : ((mat == 1) ? kb : vb);
    float scale = (mat == 0) ? SCALE_Q : 1.0f;
#pragma unroll
    for (int r = 0; r < 8; ++r) {
      size_t row = mt * 16 + r + 8 * kh;
      size_t win = row / NW;
      int tok = (int)(row % NW);
      dst[((win * NHEAD + hh) * NW + tok) * HD + d] = (__bf16)(acc[t][r] * scale);
    }
  }
}

// ---------------------------------------------------------- window attention
// One block per (window, head): 4 waves, each owns 16-row Q stripes.
// K is staged global->LDS with CDNA5 async copies (ASYNCcnt), V via VGPR
// transpose into LDS.
__global__ void attention_kernel(const __bf16* __restrict__ q,
                                 const __bf16* __restrict__ k,
                                 const __bf16* __restrict__ v,
                                 const float* __restrict__ rpb,
                                 const int* __restrict__ relidx,
                                 __bf16* __restrict__ attn_out) {
  extern __shared__ char smem[];
  __bf16* Ksh = (__bf16*)smem;                    // [NPAD][32]
  __bf16* Vtsh = Ksh + NPAD * 32;                 // [32][NPAD] (transposed V)
  float* Sall = (float*)(Vtsh + 32 * NPAD);       // 4 x [16][NPAD] f32
  __bf16* Pall = (__bf16*)(Sall + 4 * 16 * NPAD); // 4 x [16][NPAD] bf16

  int win = blockIdx.x >> 2, h = blockIdx.x & 3;
  size_t base = ((size_t)win * NHEAD + h) * NW * HD;
  const __bf16* qp = q + base;
  const __bf16* kp = k + base;
  const __bf16* vp = v + base;

  // ---- K: contiguous async global->LDS b128 copies (tracked by ASYNCcnt)
  {
    const int CHUNKS = NW * HD * 2 / 16;          // 1372 x 16B
    unsigned kbase = (unsigned)(uintptr_t)(void*)Ksh;  // low 32 bits = DS addr
    unsigned long long gbase = (unsigned long long)(uintptr_t)kp;
    for (int ch = threadIdx.x; ch < CHUNKS; ch += 128) {
      unsigned lds_off = kbase + (unsigned)ch * 16u;
      unsigned long long gaddr = gbase + (unsigned long long)ch * 16ull;
      asm volatile("global_load_async_to_lds_b128 %0, %1, off"
                   :: "v"(lds_off), "v"(gaddr)
                   : "memory");
    }
    for (int idx = threadIdx.x; idx < (NPAD - NW) * HD; idx += 128)
      Ksh[NW * HD + idx] = (__bf16)0.0f;          // zero pad rows 343..351
  }
  // ---- V: transpose into LDS through VGPRs
  for (int idx = threadIdx.x; idx < NPAD * HD; idx += 128) {
    int t = idx >> 5, d = idx & 31;
    __bf16 vv = (__bf16)0.0f;
    if (t < NW) vv = vp[(size_t)t * HD + d];
    Vtsh[d * NPAD + t] = vv;
  }
  asm volatile("s_wait_asynccnt 0x0" ::: "memory");
  __syncthreads();

  int wave = threadIdx.x >> 5, lane = threadIdx.x & 31;
  float* S = Sall + wave * 16 * NPAD;
  __bf16* P = Pall + wave * 16 * NPAD;
  int m = lane & 15, kh = lane >> 4, n = lane & 15;

  for (int mt = wave; mt < NPAD / 16; mt += 4) {
    // ---- Q fragment (zero rows past 343)
    int tok = mt * 16 + m;
    V16U qa = {};
    if (tok < NW) {
      const __bf16* qrow = qp + (size_t)tok * HD;
      qa.h[0] = *(const v8bf*)(qrow + kh * 16);
      qa.h[1] = *(const v8bf*)(qrow + kh * 16 + 8);
    }
    // ---- scores S = Q K^T + bias, spilled tile-by-tile to LDS
    for (int nt = 0; nt < NPAD / 16; ++nt) {
      v16bf bf = *(const v16bf*)(Ksh + (nt * 16 + n) * HD + kh * 16);
      v8f acc = {};
      acc = wmma_bf16(qa.v, bf, acc);
      int gn = nt * 16 + n;
      if (gn < NW) {
#pragma unroll
        for (int r = 0; r < 8; ++r) {
          int rowi = r + 8 * kh;
          int gm = mt * 16 + rowi;
          float bias = 0.0f;
          if (gm < NW) bias = rpb[(size_t)relidx[gm * NW + gn] * NHEAD + h];
          S[rowi * NPAD + gn] = acc[r] + bias;
        }
      }
    }
    // ---- row softmax (wave cooperates per row), write bf16 P
    for (int r = 0; r < 16; ++r) {
      float mx = -1e30f;
      for (int c = lane; c < NW; c += 32) mx = fmaxf(mx, S[r * NPAD + c]);
      mx = wave_max(mx);
      float sum = 0.f;
      for (int c = lane; c < NW; c += 32) {
        float e = __expf(S[r * NPAD + c] - mx);
        S[r * NPAD + c] = e;
        sum += e;
      }
      sum = wave_sum(sum);
      float inv = 1.0f / sum;
      for (int c = lane; c < NW; c += 32)
        P[r * NPAD + c] = (__bf16)(S[r * NPAD + c] * inv);
      for (int c = NW + lane; c < NPAD; c += 32) P[r * NPAD + c] = (__bf16)0.0f;
    }
    // ---- O = P @ V  (two 16-wide n-tiles cover HD=32)
    v8f o0 = {}, o1 = {};
#pragma unroll
    for (int kk = 0; kk < NPAD / 32; ++kk) {
      V16U pa;
      pa.h[0] = *(const v8bf*)(P + m * NPAD + kk * 32 + kh * 16);
      pa.h[1] = *(const v8bf*)(P + m * NPAD + kk * 32 + kh * 16 + 8);
      v16bf b0 = *(const v16bf*)(Vtsh + (size_t)n * NPAD + kk * 32 + kh * 16);
      v16bf b1 = *(const v16bf*)(Vtsh + (size_t)(16 + n) * NPAD + kk * 32 + kh * 16);
      o0 = wmma_bf16(pa.v, b0, o0);
      o1 = wmma_bf16(pa.v, b1, o1);
    }
#pragma unroll
    for (int r = 0; r < 8; ++r) {
      int gm = mt * 16 + r + 8 * kh;
      if (gm < NW) {
        size_t ob = ((size_t)win * NW + gm) * CCH + h * HD;
        attn_out[ob + n] = (__bf16)o0[r];
        attn_out[ob + 16 + n] = (__bf16)o1[r];
      }
    }
  }
}

// ------------------------------------------------- proj GEMM + reverse + res
__global__ void proj_gemm_kernel(const __bf16* __restrict__ a,
                                 const __bf16* __restrict__ w,
                                 const float* __restrict__ pb,
                                 const float* __restrict__ x,
                                 float* __restrict__ x2) {
  const int NT4 = CCH / 64;                       // 2 strips
  int wave = threadIdx.x >> 5, lane = threadIdx.x & 31;
  size_t tile = (size_t)blockIdx.x * 4 + wave;
  int nt4 = (int)(tile % NT4);
  size_t mt = tile / NT4;
  if (mt >= MTILES) return;
  int n = lane & 15, kh = lane >> 4;
  v8f acc[4];
#pragma unroll
  for (int t = 0; t < 4; ++t) {
    float bv = pb[(nt4 * 4 + t) * 16 + n];
#pragma unroll
    for (int r = 0; r < 8; ++r) acc[t][r] = bv;
  }
  gemm_strip4<CCH>(a, w, mt, nt4 * 4, acc);
#pragma unroll
  for (int t = 0; t < 4; ++t) {
    int col = (nt4 * 4 + t) * 16 + n;
#pragma unroll
    for (int r = 0; r < 8; ++r) {
      size_t row = mt * 16 + r + 8 * kh;
      size_t win = row / NW;
      int tok = (int)(row % NW);
      int bb = (int)(win >> 6), rem = (int)(win & 63);
      int wh = rem >> 4, ww = (rem >> 2) & 3, wd = rem & 3;
      int i = tok / 49, j = (tok / 7) % 7, kz = tok % 7;
      int ah = (wh * 7 + i + 3) % 28;             // roll(+3) scatter
      int aw = (ww * 7 + j + 3) % 28;
      int ad = (wd * 7 + kz + 3) % 28;
      size_t orig = ((size_t)ah * 28 + aw) * 28 + ad;
      size_t idx = ((size_t)bb * TOKS_B + orig) * CCH + col;
      x2[idx] = x[idx] + acc[t][r];
    }
  }
}

// ------------------------------------------------------------ MLP1 (+GELU)
__global__ void mlp1_gemm_kernel(const __bf16* __restrict__ a,
                                 const __bf16* __restrict__ w,
                                 const float* __restrict__ b1,
                                 __bf16* __restrict__ hid) {
  const int NT4 = HIDDEN / 64;                    // 8 strips
  int wave = threadIdx.x >> 5, lane = threadIdx.x & 31;
  size_t tile = (size_t)blockIdx.x * 4 + wave;
  int nt4 = (int)(tile % NT4);
  size_t mt = tile / NT4;
  if (mt >= MTILES) return;
  int n = lane & 15, kh = lane >> 4;
  v8f acc[4];
#pragma unroll
  for (int t = 0; t < 4; ++t) {
    float bv = b1[(nt4 * 4 + t) * 16 + n];
#pragma unroll
    for (int r = 0; r < 8; ++r) acc[t][r] = bv;
  }
  gemm_strip4<CCH>(a, w, mt, nt4 * 4, acc);
#pragma unroll
  for (int t = 0; t < 4; ++t) {
    int col = (nt4 * 4 + t) * 16 + n;
#pragma unroll
    for (int r = 0; r < 8; ++r) {
      size_t row = mt * 16 + r + 8 * kh;
      float g = acc[t][r];
      g = 0.5f * g * (1.0f + erff(g * 0.70710678118654752f));  // exact GELU
      hid[row * HIDDEN + col] = (__bf16)g;
    }
  }
}

// ------------------------------------------------------------ MLP2 (+res)
__global__ void mlp2_gemm_kernel(const __bf16* __restrict__ a,
                                 const __bf16* __restrict__ w,
                                 const float* __restrict__ b2,
                                 const float* __restrict__ x2,
                                 float* __restrict__ out) {
  const int NT4 = CCH / 64;                       // 2 strips
  int wave = threadIdx.x >> 5, lane = threadIdx.x & 31;
  size_t tile = (size_t)blockIdx.x * 4 + wave;
  int nt4 = (int)(tile % NT4);
  size_t mt = tile / NT4;
  if (mt >= MTILES) return;
  int n = lane & 15, kh = lane >> 4;
  v8f acc[4];
#pragma unroll
  for (int t = 0; t < 4; ++t) {
    float bv = b2[(nt4 * 4 + t) * 16 + n];
#pragma unroll
    for (int r = 0; r < 8; ++r) acc[t][r] = bv;
  }
  gemm_strip4<HIDDEN>(a, w, mt, nt4 * 4, acc);
#pragma unroll
  for (int t = 0; t < 4; ++t) {
    int col = (nt4 * 4 + t) * 16 + n;
#pragma unroll
    for (int r = 0; r < 8; ++r) {
      size_t row = mt * 16 + r + 8 * kh;
      size_t idx = row * CCH + col;
      out[idx] = x2[idx] + acc[t][r];
    }
  }
}

// ---------------------------------------------------------------- launcher
extern "C" void kernel_launch(void* const* d_in, const int* in_sizes, int n_in,
                              void* d_out, int out_size, void* d_ws, size_t ws_size,
                              hipStream_t stream) {
  const float* x       = (const float*)d_in[0];
  const float* norm1_w = (const float*)d_in[1];
  const float* norm1_b = (const float*)d_in[2];
  const float* qkv_w   = (const float*)d_in[3];
  const float* qkv_b   = (const float*)d_in[4];
  const float* rpb     = (const float*)d_in[5];
  const float* proj_w  = (const float*)d_in[6];
  const float* proj_b  = (const float*)d_in[7];
  const float* norm2_w = (const float*)d_in[8];
  const float* norm2_b = (const float*)d_in[9];
  const float* mlp_w1  = (const float*)d_in[10];
  const float* mlp_b1  = (const float*)d_in[11];
  const float* mlp_w2  = (const float*)d_in[12];
  const float* mlp_b2  = (const float*)d_in[13];
  const int*   relidx  = (const int*)d_in[14];
  float* out = (float*)d_out;

  char* ws = (char*)d_ws;
  __bf16* xw    = (__bf16*)ws; ws += (size_t)TOKENS * CCH * 2;       // 11.2 MB
  __bf16* wqkvb = (__bf16*)ws; ws += 384 * CCH * 2;
  __bf16* wprojb= (__bf16*)ws; ws += CCH * CCH * 2;
  __bf16* w1b   = (__bf16*)ws; ws += HIDDEN * CCH * 2;
  __bf16* w2b   = (__bf16*)ws; ws += CCH * HIDDEN * 2;
  __bf16* qb    = (__bf16*)ws; ws += (size_t)TOKENS * CCH * 2;
  __bf16* kb    = (__bf16*)ws; ws += (size_t)TOKENS * CCH * 2;
  __bf16* vb    = (__bf16*)ws; ws += (size_t)TOKENS * CCH * 2;
  __bf16* attn  = (__bf16*)ws; ws += (size_t)TOKENS * CCH * 2;
  float*  x2    = (float*)ws;  ws += (size_t)TOKENS * CCH * 4;
  __bf16* ln2o  = (__bf16*)ws; ws += (size_t)TOKENS * CCH * 2;
  __bf16* hid   = (__bf16*)ws; ws += (size_t)TOKENS * HIDDEN * 2;

  // weights -> bf16
  cvt_f32_bf16_kernel<<<(384 * CCH + 255) / 256, 256, 0, stream>>>(qkv_w, wqkvb, 384 * CCH);
  cvt_f32_bf16_kernel<<<(CCH * CCH + 255) / 256, 256, 0, stream>>>(proj_w, wprojb, CCH * CCH);
  cvt_f32_bf16_kernel<<<(HIDDEN * CCH + 255) / 256, 256, 0, stream>>>(mlp_w1, w1b, HIDDEN * CCH);
  cvt_f32_bf16_kernel<<<(CCH * HIDDEN + 255) / 256, 256, 0, stream>>>(mlp_w2, w2b, CCH * HIDDEN);

  // LN1 + shift + window partition (wave per token)
  ln1_partition_kernel<<<TOKENS / 4, 128, 0, stream>>>(x, norm1_w, norm1_b, xw);

  // QKV projection (wave per 16x64 strip)
  qkv_gemm_kernel<<<(MTILES * 6) / 4, 128, 0, stream>>>(xw, wqkvb, qkv_b, qb, kb, vb);

  // attention: block per (window, head)
  const int SMEM_ATTN = (NPAD * 32 * 2) * 2 + 4 * 16 * NPAD * 4 + 4 * 16 * NPAD * 2; // 180224
  (void)hipFuncSetAttribute(reinterpret_cast<const void*>(attention_kernel),
                            hipFuncAttributeMaxDynamicSharedMemorySize, SMEM_ATTN);
  attention_kernel<<<NWIN * NHEAD, 128, SMEM_ATTN, stream>>>(qb, kb, vb, rpb, relidx, attn);

  // proj + window reverse + roll + residual
  proj_gemm_kernel<<<(MTILES * 2) / 4, 128, 0, stream>>>(attn, wprojb, proj_b, x, x2);

  // LN2
  ln2_kernel<<<TOKENS / 4, 128, 0, stream>>>(x2, norm2_w, norm2_b, ln2o);

  // MLP
  mlp1_gemm_kernel<<<(MTILES * 8) / 4, 128, 0, stream>>>(ln2o, w1b, mlp_b1, hid);
  mlp2_gemm_kernel<<<(MTILES * 2) / 4, 128, 0, stream>>>(hid, w2b, mlp_b2, x2, out);
}
